// AttentionLayer_55353538511035
// MI455X (gfx1250) — matrix operand order repaired and checked by reading, not compiled, not run on hardware
//
#include <hip/hip_runtime.h>

typedef __attribute__((ext_vector_type(16))) __bf16 v16bf;
typedef __attribute__((ext_vector_type(8)))  float  v8f;

#define GAT_N     8192
#define GAT_INF   512
#define GAT_OUTF  256
#define NEG_BIG   (-9.0e15f)

// K index inside a 16-bit WMMA A fragment (16x32): element e, half = lane>=16
__device__ __forceinline__ int a_kmap(int e, int hi) {
    return e + hi * 16;
}

// Swizzled bf16 matrix layout for WMMA-B operands:
//   addr(k, n) = ((k>>5)*OUT_F + n)*32 + (k&31)
// A lane (n = col, hi) then reads 16 contiguous bf16 (32 B) per fragment.

// ---------------------------------------------------------------------------
// Kernel 0: convert W (fp32 [IN_F, OUT_F]) into swizzled bf16.
// ---------------------------------------------------------------------------
__global__ void __launch_bounds__(256)
gat_wswz_kernel(const float* __restrict__ W, __bf16* __restrict__ Wswz) {
    const int idx = blockIdx.x * 256 + threadIdx.x;   // 0 .. IN_F*OUT_F-1
    const int k = idx >> 8;           // OUT_F == 256
    const int n = idx & 255;
    Wswz[(((k >> 5) * GAT_OUTF + n) << 5) + (k & 31)] = (__bf16)W[idx];
}

// ---------------------------------------------------------------------------
// Kernel 1: Wh = h @ W  (bf16 WMMA, fp32 accum). Emits swizzled bf16 Wh and
// per-row attention scalars attn_src = Wh@a_src, attn_dst = Wh@a_dst.
// Block = 128 threads (4 waves): 16 rows x 256 cols; wave owns 16x64.
// ---------------------------------------------------------------------------
__global__ void __launch_bounds__(128)
gat_wh_kernel(const float* __restrict__ h,       // [N, IN_F]
              const __bf16* __restrict__ Wswz,   // swizzled [IN_F, OUT_F]
              const float* __restrict__ a_src,   // [OUT_F]
              const float* __restrict__ a_dst,   // [OUT_F]
              __bf16* __restrict__ Whb,          // swizzled [N, OUT_F]
              float* __restrict__ attn_src,      // [N]
              float* __restrict__ attn_dst)      // [N]
{
    const int rowBlock = blockIdx.x * 16;
    const int lane = threadIdx.x & 31;
    const int wave = threadIdx.x >> 5;   // 0..3
    const int hi   = lane >> 4;
    const int l16  = lane & 15;

    __shared__ float redS[16], redD[16];
    if (threadIdx.x < 16) { redS[threadIdx.x] = 0.f; redD[threadIdx.x] = 0.f; }
    __syncthreads();

    v8f acc[4] = {};
    const float* hrow = h + (size_t)(rowBlock + l16) * GAT_INF;

    for (int kb = 0; kb < GAT_INF; kb += 32) {
        // A fragment: h[row][kb + kmap(e,hi)] -> bf16
        v16bf afrag;
#pragma unroll
        for (int e = 0; e < 16; ++e)
            afrag[e] = (__bf16)hrow[kb + a_kmap(e, hi)];

        // B fragments: contiguous 32-byte vector loads from swizzled W
#pragma unroll
        for (int t = 0; t < 4; ++t) {
            const int col = wave * 64 + t * 16 + l16;
            const v16bf bfrag =
                *(const v16bf*)(Wswz + (((size_t)(kb >> 5) * GAT_OUTF + col) << 5) + hi * 16);
            acc[t] = __builtin_amdgcn_wmma_f32_16x16x32_bf16(
                false, afrag, false, bfrag, (short)0, acc[t], false, false);
        }
    }

    // Epilogue: swizzled bf16 Wh store + LDS reduction of attention scalars.
#pragma unroll
    for (int t = 0; t < 4; ++t) {
        const int col = wave * 64 + t * 16 + l16;
        const float as = a_src[col];
        const float ad = a_dst[col];
#pragma unroll
        for (int r = 0; r < 8; ++r) {
            const int row = rowBlock + r + hi * 8;
            const float v = acc[t][r];
            Whb[(((row >> 5) * GAT_OUTF + col) << 5) + (row & 31)] = (__bf16)v;
            atomicAdd(&redS[r + hi * 8], v * as);
            atomicAdd(&redD[r + hi * 8], v * ad);
        }
    }
    __syncthreads();
    if (threadIdx.x < 16) {
        attn_src[rowBlock + threadIdx.x] = redS[threadIdx.x];
        attn_dst[rowBlock + threadIdx.x] = redD[threadIdx.x];
    }
}

// ---------------------------------------------------------------------------
// Kernel 2: fused masked-softmax attention (flash style), 64-wide j chunks.
// Block = 256 threads (8 waves) owns a 16-row slab. One row per half-wave:
// shuffle-reduced online softmax, bf16 P tile stored in LDS in A-fragment
// order, Wh consumed from the swizzled layout as 32-byte B fragments.
// ---------------------------------------------------------------------------
__global__ void __launch_bounds__(256)
gat_attn_kernel(const int* __restrict__ adj,        // [N, N]
                const __bf16* __restrict__ Whb,     // swizzled [N, OUT_F]
                const float* __restrict__ attn_src, // [N]
                const float* __restrict__ attn_dst, // [N]
                float* __restrict__ out)            // [N, OUT_F]
{
    const int rowBlock = blockIdx.x * 16;
    const int tid  = threadIdx.x;
    const int lane = tid & 31;
    const int wave = tid >> 5;       // 0..7 -> 32-col slice
    const int hi   = lane >> 4;
    const int l16  = lane & 15;
    const int r    = tid >> 4;       // row owned by this half-wave (0..15)
    const int ln   = tid & 15;       // lane within half-wave

    __shared__ __bf16 sP[16 * 64];   // P tile, A-fragment order: row*64 + kk + hi*16 + e
    __shared__ float  mrow[16], lrow[16], srow[16], ssrc[16];

    if (tid < 16) {
        mrow[tid] = -__builtin_inff();
        lrow[tid] = 0.f;
        ssrc[tid] = attn_src[rowBlock + tid];
    }
    __syncthreads();

    v8f acc[2] = {};
    const int*   adjrow = adj + (size_t)(rowBlock + r) * GAT_N;

    for (int jb = 0; jb < GAT_N; jb += 64) {
        // ---- Stage A: scores + parallel online softmax (one row/half-wave) --
        const int j0 = jb + 4 * ln;
        const int4   a4 = *(const int4*)(adjrow + j0);
        const float4 d4 = *(const float4*)(attn_dst + j0);
        if (jb + 64 < GAT_N)
            __builtin_prefetch(adjrow + j0 + 64, 0, 1);

        const float src = ssrc[r];
        float s[4];
        {
            const float dv[4] = {d4.x, d4.y, d4.z, d4.w};
            const int   av[4] = {a4.x, a4.y, a4.z, a4.w};
#pragma unroll
            for (int q = 0; q < 4; ++q) {
                float x = src + dv[q];
                x = x > 0.f ? x : 0.2f * x;
                s[q] = (av[q] > 0) ? x : NEG_BIG;
            }
        }
        float mloc = fmaxf(fmaxf(s[0], s[1]), fmaxf(s[2], s[3]));
#pragma unroll
        for (int msk = 1; msk < 16; msk <<= 1)
            mloc = fmaxf(mloc, __shfl_xor(mloc, msk));
        const float mold = mrow[r];
        const float mc = fmaxf(mold, mloc);
        const float sc = __expf(mold - mc);
        float p[4], lloc = 0.f;
#pragma unroll
        for (int q = 0; q < 4; ++q) { p[q] = __expf(s[q] - mc); lloc += p[q]; }
#pragma unroll
        for (int msk = 1; msk < 16; msk <<= 1)
            lloc += __shfl_xor(lloc, msk);
        if (ln == 0) {
            mrow[r] = mc;
            lrow[r] = lrow[r] * sc + lloc;
            srow[r] = sc;
        }
        // Store P in A-fragment order: j -> (kk = j&32, hi2, e)
#pragma unroll
        for (int q = 0; q < 4; ++q) {
            const int j   = 4 * ln + q;          // 0..63
            const int jj  = j & 31;
            const int hi2 = jj >> 4;
            const int e   = jj & 15;
            sP[r * 64 + (j & 32) + hi2 * 16 + e] = (__bf16)p[q];
        }
        __syncthreads();

        // ---- Stage B: rescale accumulators, acc += P @ Wh_chunk (2 K-halves)
        const v16bf a0 = *(const v16bf*)&sP[l16 * 64 + 0  + hi * 16];
        const v16bf a1 = *(const v16bf*)&sP[l16 * 64 + 32 + hi * 16];
#pragma unroll
        for (int t = 0; t < 2; ++t) {
            const int col = wave * 32 + t * 16 + l16;
            const __bf16* bp =
                Whb + (((size_t)(jb >> 5) * GAT_OUTF + col) << 5) + hi * 16;
            const v16bf b0 = *(const v16bf*)bp;
            const v16bf b1 = *(const v16bf*)(bp + GAT_OUTF * 32);  // next 32-row group
#pragma unroll
            for (int rr = 0; rr < 8; ++rr)
                acc[t][rr] *= srow[rr + hi * 8];
            acc[t] = __builtin_amdgcn_wmma_f32_16x16x32_bf16(
                false, a0, false, b0, (short)0, acc[t], false, false);
            acc[t] = __builtin_amdgcn_wmma_f32_16x16x32_bf16(
                false, a1, false, b1, (short)0, acc[t], false, false);
        }
        __syncthreads();
    }

    // ---- Epilogue: normalize by softmax denominator, write fp32 output ----
#pragma unroll
    for (int t = 0; t < 2; ++t) {
        const int col = wave * 32 + t * 16 + l16;
#pragma unroll
        for (int rr = 0; rr < 8; ++rr) {
            const int row = rr + hi * 8;
            out[(size_t)(rowBlock + row) * GAT_OUTF + col] = acc[t][rr] / lrow[row];
        }
    }
}

extern "C" void kernel_launch(void* const* d_in, const int* in_sizes, int n_in,
                              void* d_out, int out_size, void* d_ws, size_t ws_size,
                              hipStream_t stream) {
    const float* h     = (const float*)d_in[0];  // [N, IN_F]
    const int*   adj   = (const int*)  d_in[1];  // [N, N]
    const float* W     = (const float*)d_in[2];  // [IN_F, OUT_F]
    const float* a_src = (const float*)d_in[3];  // [OUT_F]
    const float* a_dst = (const float*)d_in[4];  // [OUT_F]
    float* out = (float*)d_out;                  // [N, OUT_F]

    // Workspace: swizzled bf16 Wh (4 MB) | swizzled bf16 W (256 KB) | attn vecs
    __bf16* Whb  = (__bf16*)d_ws;
    __bf16* Wswz = Whb + (size_t)GAT_N * GAT_OUTF;
    float*  attn_src = (float*)(Wswz + (size_t)GAT_INF * GAT_OUTF);
    float*  attn_dst = attn_src + GAT_N;

    gat_wswz_kernel<<<(GAT_INF * GAT_OUTF) / 256, 256, 0, stream>>>(W, Wswz);
    gat_wh_kernel<<<GAT_N / 16, 128, 0, stream>>>(h, Wswz, a_src, a_dst,
                                                  Whb, attn_src, attn_dst);
    gat_attn_kernel<<<GAT_N / 16, 256, 0, stream>>>(adj, Whb, attn_src, attn_dst, out);
}
